// GNN_66383014527243
// MI455X (gfx1250) — compile-verified, matrix operand order checked by reading, not run on hardware
//
#include <hip/hip_runtime.h>
#include <math.h>

// ---------------------------------------------------------------------------
// GATv2 GNN forward for MI455X (gfx1250, wave32, WMMA).
// Big GEMMs run as f16 v_wmma_f32_16x16x32_f16 with f32 accumulation.
// Wave tile = 16x64 (4 WMMAs per 32-K step) -- spill-free register footprint.
// ---------------------------------------------------------------------------

#define N_NODES 25000
#define N_PAD   25024      // multiple of 16 (M padded to WMMA tile height)
#define E_EDGES 50000
#define E_PAD   50016      // multiple of 16
#define F_NODEC 7
#define F_EDGEC 2
#define HEADS   4
#define C_DIM   512
#define HC_DIM  2048

typedef __attribute__((ext_vector_type(16))) _Float16 v16h;
typedef __attribute__((ext_vector_type(8)))  _Float16 v8h;
typedef __attribute__((ext_vector_type(8)))  float    v8f;

// ----------------------------- helpers -------------------------------------

__device__ __forceinline__ float atomicMaxFloat(float* addr, float v) {
    // order-preserving int punning; addr initialized to -inf
    if (v >= 0.0f)
        return __int_as_float(atomicMax((int*)addr, __float_as_int(v)));
    else
        return __uint_as_float(atomicMin((unsigned int*)addr, __float_as_uint(v)));
}

// ------------------------ weight convert / transpose ------------------------
// W: f32 [K][Nc] row-major  ->  Wt: f16 [Nc][K] row-major (B^T panel layout
// so each WMMA B-fragment lane loads contiguous K runs as b128).
__global__ void cvt_transpose_f16_kernel(const float* __restrict__ W,
                                         _Float16* __restrict__ Wt,
                                         int K, int Nc) {
    long long id = (long long)blockIdx.x * blockDim.x + threadIdx.x;
    if (id >= (long long)K * Nc) return;
    int n = (int)(id / K);
    int k = (int)(id % K);
    Wt[(size_t)n * K + k] = (_Float16)W[(size_t)k * Nc + n];
}

// ------------------------------- encoders -----------------------------------
__global__ void node_encoder_kernel(const float* __restrict__ x,
                                    const float* __restrict__ Wn,
                                    const float* __restrict__ bn,
                                    _Float16* __restrict__ h) {
    long long id = (long long)blockIdx.x * blockDim.x + threadIdx.x;
    if (id >= (long long)N_PAD * C_DIM) return;
    int n = (int)(id / C_DIM);
    int c = (int)(id % C_DIM);
    float acc = 0.0f;
    if (n < N_NODES) {
        acc = bn[c];
        #pragma unroll
        for (int f = 0; f < F_NODEC; ++f)
            acc += x[(size_t)n * F_NODEC + f] * Wn[f * C_DIM + c];
        acc = fmaxf(acc, 0.0f);
    }
    h[(size_t)n * C_DIM + c] = (_Float16)acc;   // pad rows -> 0
}

__global__ void edge_encoder_kernel(const float* __restrict__ ea,
                                    const float* __restrict__ We,
                                    const float* __restrict__ be,
                                    _Float16* __restrict__ eah) {
    long long id = (long long)blockIdx.x * blockDim.x + threadIdx.x;
    if (id >= (long long)E_PAD * C_DIM) return;
    int e = (int)(id / C_DIM);
    int c = (int)(id % C_DIM);
    float acc = 0.0f;
    if (e < E_EDGES) {
        acc = be[c];
        #pragma unroll
        for (int f = 0; f < F_EDGEC; ++f)
            acc += ea[(size_t)e * F_EDGEC + f] * We[f * C_DIM + c];
        acc = fmaxf(acc, 0.0f);
    }
    eah[(size_t)e * C_DIM + c] = (_Float16)acc;  // pad rows -> 0
}

// ------------------------------ WMMA GEMM -----------------------------------
// C[M,Nc] (f32) = A[M,K] (f16, row-major) @ Bt[Nc,K]^T (f16) + bias, opt ReLU.
// Block = 128 threads = 4 waves. Each wave owns a 16x64 tile (4 N-subtiles,
// 4 v_wmma_f32_16x16x32_f16 per 32-K step). All 5 fragments are loaded before
// the WMMA group so the scheduler can stagger s_wait_loadcnt across them.
// ~130 live VGPRs incl. the compiler's 2-deep pipeline -> no scratch spills.
// Fragment layout per CDNA5 ISA 7.12.2:
//   A 16x32 f16: lanes 0-15 rows M=0..15 / lanes 16-31 same rows, K offset +8;
//   per lane K runs {k..k+7} and {k+16..k+23} -> two contiguous b128 loads.
//   B 32x16 f16 mirrors A with lane = output column (hence the B^T panels).
__global__ void __launch_bounds__(128)
wmma_gemm_kernel(const _Float16* __restrict__ A,
                 const _Float16* __restrict__ Bt,
                 const float* __restrict__ bias,
                 float* __restrict__ C,
                 int M, int Nc, int K, int relu) {
    const int lane = threadIdx.x & 31;
    const int wave = threadIdx.x >> 5;
    const int m0 = blockIdx.x * 16;
    const int n0 = blockIdx.y * 256 + wave * 64;
    if (m0 >= M || n0 >= Nc) return;

    const int row  = lane & 15;
    const int ksel = (lane >> 4) << 3;      // +8 K offset for upper half-wave

    v8f acc[4] = {};
    const _Float16* ap = A + (size_t)(m0 + row) * K + ksel;

    for (int k = 0; k < K; k += 32) {
        v16h a;
        *((v8h*)&a)     = *(const v8h*)(ap + k);
        *((v8h*)&a + 1) = *(const v8h*)(ap + k + 16);

        v16h b[4];
        #pragma unroll
        for (int t = 0; t < 4; ++t) {
            const _Float16* bp = Bt + (size_t)(n0 + t * 16 + row) * K + ksel + k;
            *((v8h*)&b[t])     = *(const v8h*)(bp);
            *((v8h*)&b[t] + 1) = *(const v8h*)(bp + 16);
        }
        if (k + 32 < K)                     // near-scope prefetch of next panel
            __builtin_prefetch(ap + k + 32, 0, 3);

        #pragma unroll
        for (int t = 0; t < 4; ++t)
            acc[t] = __builtin_amdgcn_wmma_f32_16x16x32_f16(
                false, a, false, b[t], (short)0, acc[t], false, false);
    }

    // D layout: lane n = lane&15, rows m = vgpr + 8*(lane>>4)
    const int mrow = (lane >> 4) * 8;
    #pragma unroll
    for (int t = 0; t < 4; ++t) {
        int n = n0 + t * 16 + (lane & 15);
        float bv = bias ? bias[n] : 0.0f;
        #pragma unroll
        for (int i = 0; i < 8; ++i) {
            float v = acc[t][i] + bv;
            if (relu) v = fmaxf(v, 0.0f);
            C[(size_t)(m0 + mrow + i) * Nc + n] = v;
        }
    }
}

// ------------------------- softmax state init -------------------------------
__global__ void init_softmax_kernel(float* __restrict__ mx, float* __restrict__ den) {
    int id = blockIdx.x * blockDim.x + threadIdx.x;
    if (id >= N_NODES * HEADS) return;
    mx[id]  = -INFINITY;
    den[id] = 0.0f;
}

__global__ void init_out_kernel(float* __restrict__ out,
                                const float* __restrict__ conv_bias) {
    long long id = (long long)blockIdx.x * blockDim.x + threadIdx.x;
    if (id >= (long long)N_PAD * HC_DIM) return;
    out[id] = conv_bias[(int)(id % HC_DIM)];
}

// ----------------------- edge logits + segment max --------------------------
// block = edge, wave = head; lanes reduce the 512-wide attention dot product
// with float4 (b128) loads: lane q covers c = q*4 .. q*4+3.
__global__ void edge_logits_kernel(const float* __restrict__ xl,
                                   const float* __restrict__ xr,
                                   const float* __restrict__ ee,
                                   const int* __restrict__ ei,
                                   const float* __restrict__ att,
                                   float* __restrict__ logits,
                                   float* __restrict__ mx) {
    const int e    = blockIdx.x;
    const int h    = threadIdx.x >> 5;
    const int lane = threadIdx.x & 31;
    const int src  = ei[e];
    const int dst  = ei[E_EDGES + e];

    const float4* pl = (const float4*)(xl + (size_t)src * HC_DIM + h * C_DIM);
    const float4* pr = (const float4*)(xr + (size_t)dst * HC_DIM + h * C_DIM);
    const float4* pe = (const float4*)(ee + (size_t)e   * HC_DIM + h * C_DIM);
    const float4* pa = (const float4*)(att + h * C_DIM);

    float acc = 0.0f;
    #pragma unroll
    for (int q = lane; q < C_DIM / 4; q += 32) {
        float4 l = pl[q], r = pr[q], ev = pe[q], a = pa[q];
        float m, s;
        m = l.x + r.x + ev.x; s = (m > 0.0f) ? m : 0.2f * m; acc += s * a.x;
        m = l.y + r.y + ev.y; s = (m > 0.0f) ? m : 0.2f * m; acc += s * a.y;
        m = l.z + r.z + ev.z; s = (m > 0.0f) ? m : 0.2f * m; acc += s * a.z;
        m = l.w + r.w + ev.w; s = (m > 0.0f) ? m : 0.2f * m; acc += s * a.w;
    }
    #pragma unroll
    for (int o = 16; o > 0; o >>= 1) acc += __shfl_down(acc, o, 32);
    if (lane == 0) {
        logits[e * HEADS + h] = acc;
        atomicMaxFloat(&mx[dst * HEADS + h], acc);
    }
}

// ------------------------- exp + segment sum --------------------------------
__global__ void edge_exp_kernel(const float* __restrict__ logits,
                                const float* __restrict__ mx,
                                const int* __restrict__ ei,
                                float* __restrict__ ex,
                                float* __restrict__ den) {
    int id = blockIdx.x * blockDim.x + threadIdx.x;
    if (id >= E_EDGES * HEADS) return;
    int e = id / HEADS, h = id % HEADS;
    int dst = ei[E_EDGES + e];
    float v = __expf(logits[id] - mx[dst * HEADS + h]);
    ex[id] = v;
    atomicAdd(&den[dst * HEADS + h], v);
}

// --------------------- alpha-weighted scatter aggregation -------------------
// block = edge, 256 threads; float4 source loads, scalar f32 atomics out
// (out accumulator ~200 MB: L2-resident, atomics complete at L2).
__global__ void edge_aggregate_kernel(const float* __restrict__ xl,
                                      const int* __restrict__ ei,
                                      const float* __restrict__ ex,
                                      const float* __restrict__ den,
                                      float* __restrict__ out) {
    const int e   = blockIdx.x;
    const int src = ei[e];
    const int dst = ei[E_EDGES + e];
    float alpha[HEADS];
    #pragma unroll
    for (int h = 0; h < HEADS; ++h)
        alpha[h] = ex[e * HEADS + h] / (den[dst * HEADS + h] + 1e-16f);
    const float4* ps = (const float4*)(xl + (size_t)src * HC_DIM);
    float*        pd = out + (size_t)dst * HC_DIM;
    #pragma unroll
    for (int q = threadIdx.x; q < HC_DIM / 4; q += 256) {
        float4 v = ps[q];
        float  al = alpha[q >> 7];          // (q*4) >> 9
        int    c  = q * 4;
        atomicAdd(&pd[c + 0], al * v.x);
        atomicAdd(&pd[c + 1], al * v.y);
        atomicAdd(&pd[c + 2], al * v.z);
        atomicAdd(&pd[c + 3], al * v.w);
    }
}

// --------------------------- f32 -> f16 convert -----------------------------
__global__ void cvt_f16_kernel(const float* __restrict__ src,
                               _Float16* __restrict__ dst, long long n4) {
    long long id = (long long)blockIdx.x * blockDim.x + threadIdx.x;
    if (id >= n4) return;
    float4 v = ((const float4*)src)[id];
    _Float16 h4[4] = {(_Float16)v.x, (_Float16)v.y, (_Float16)v.z, (_Float16)v.w};
    ((ushort2*)dst)[id * 2 + 0] = *(ushort2*)&h4[0];
    ((ushort2*)dst)[id * 2 + 1] = *(ushort2*)&h4[2];
}

// ----------------------- tiny decoder head + sigmoid ------------------------
__global__ void decoder2_kernel(const float* __restrict__ t1,   // [N,512] relu'd
                                const float* __restrict__ Wd2,  // [512][6]
                                const float* __restrict__ bd2,
                                float* __restrict__ pred) {
    int id = blockIdx.x * blockDim.x + threadIdx.x;
    if (id >= N_NODES * 6) return;
    int n = id / 6, j = id % 6;
    const float* t = t1 + (size_t)n * C_DIM;
    float acc = bd2[j];
    for (int c = 0; c < C_DIM; ++c)
        acc += t[c] * Wd2[c * 6 + j];
    pred[id] = 1.0f / (1.0f + __expf(-acc));
}

// ---------------------------------------------------------------------------

static inline size_t alignup(size_t x) { return (x + 255) & ~(size_t)255; }

extern "C" void kernel_launch(void* const* d_in, const int* in_sizes, int n_in,
                              void* d_out, int out_size, void* d_ws, size_t ws_size,
                              hipStream_t stream) {
    const float* x         = (const float*)d_in[0];
    const float* edge_attr = (const float*)d_in[1];
    const int*   edge_idx  = (const int*)  d_in[2];
    const float* W_node    = (const float*)d_in[3];
    const float* b_node    = (const float*)d_in[4];
    const float* W_edge    = (const float*)d_in[5];
    const float* b_edge    = (const float*)d_in[6];
    const float* W_l       = (const float*)d_in[7];
    const float* b_l       = (const float*)d_in[8];
    const float* W_r       = (const float*)d_in[9];
    const float* b_r       = (const float*)d_in[10];
    const float* W_le      = (const float*)d_in[11];
    const float* att       = (const float*)d_in[12];
    const float* conv_bias = (const float*)d_in[13];
    const float* W_d1      = (const float*)d_in[14];
    const float* b_d1      = (const float*)d_in[15];
    const float* W_d2      = (const float*)d_in[16];
    const float* b_d2      = (const float*)d_in[17];
    float* pred = (float*)d_out;

    // ---- workspace layout (peak ~915 MB; xr is reused as `out`, ee region
    //      is reused for the decoder intermediates once edges are consumed) ----
    char* ws = (char*)d_ws;
    size_t off = 0;
    _Float16* h_f16  = (_Float16*)(ws + off); off += alignup((size_t)N_PAD * C_DIM * 2);
    _Float16* ea_f16 = (_Float16*)(ws + off); off += alignup((size_t)E_PAD * C_DIM * 2);
    _Float16* Wl_t   = (_Float16*)(ws + off); off += alignup((size_t)HC_DIM * C_DIM * 2);
    _Float16* Wr_t   = (_Float16*)(ws + off); off += alignup((size_t)HC_DIM * C_DIM * 2);
    _Float16* Wle_t  = (_Float16*)(ws + off); off += alignup((size_t)HC_DIM * C_DIM * 2);
    _Float16* Wd1_t  = (_Float16*)(ws + off); off += alignup((size_t)C_DIM * HC_DIM * 2);
    float*    xl     = (float*)   (ws + off); off += alignup((size_t)N_PAD * HC_DIM * 4);
    float*    xr     = (float*)   (ws + off); off += alignup((size_t)N_PAD * HC_DIM * 4);
    char*     ee_rg  =            (ws + off); off += alignup((size_t)E_PAD * HC_DIM * 4);
    float*    logits = (float*)   (ws + off); off += alignup((size_t)E_EDGES * HEADS * 4);
    float*    ex     = (float*)   (ws + off); off += alignup((size_t)E_EDGES * HEADS * 4);
    float*    mx     = (float*)   (ws + off); off += alignup((size_t)N_NODES * HEADS * 4);
    float*    den    = (float*)   (ws + off); off += alignup((size_t)N_NODES * HEADS * 4);

    float*    ee      = (float*)ee_rg;                                   // pass A
    float*    out     = xr;                                              // after pass A
    _Float16* out_f16 = (_Float16*)ee_rg;                                // after aggregation
    float*    t1      = (float*)(ee_rg + alignup((size_t)N_PAD * HC_DIM * 2));

    const int TB = 256;
    dim3 blk(TB);

    // 1) weight panels -> f16 transposed [Nc][K]
    {
        long long n1 = (long long)C_DIM * HC_DIM;
        dim3 g((unsigned)((n1 + TB - 1) / TB));
        cvt_transpose_f16_kernel<<<g, blk, 0, stream>>>(W_l,  Wl_t,  C_DIM, HC_DIM);
        cvt_transpose_f16_kernel<<<g, blk, 0, stream>>>(W_r,  Wr_t,  C_DIM, HC_DIM);
        cvt_transpose_f16_kernel<<<g, blk, 0, stream>>>(W_le, Wle_t, C_DIM, HC_DIM);
        cvt_transpose_f16_kernel<<<g, blk, 0, stream>>>(W_d1, Wd1_t, HC_DIM, C_DIM);
    }

    // 2) encoders (f16 activations, M-padded/zero-filled)
    {
        long long n1 = (long long)N_PAD * C_DIM;
        node_encoder_kernel<<<dim3((unsigned)((n1 + TB - 1) / TB)), blk, 0, stream>>>(
            x, W_node, b_node, h_f16);
        long long n2 = (long long)E_PAD * C_DIM;
        edge_encoder_kernel<<<dim3((unsigned)((n2 + TB - 1) / TB)), blk, 0, stream>>>(
            edge_attr, W_edge, b_edge, ea_f16);
    }

    // 3) the three big WMMA GEMMs: xl, xr (with bias), ee (no bias)
    {
        dim3 g(N_PAD / 16, HC_DIM / 256), b128(128);
        wmma_gemm_kernel<<<g, b128, 0, stream>>>(h_f16, Wl_t, b_l, xl,
                                                 N_PAD, HC_DIM, C_DIM, 0);
        wmma_gemm_kernel<<<g, b128, 0, stream>>>(h_f16, Wr_t, b_r, xr,
                                                 N_PAD, HC_DIM, C_DIM, 0);
        dim3 ge(E_PAD / 16, HC_DIM / 256);
        wmma_gemm_kernel<<<ge, b128, 0, stream>>>(ea_f16, Wle_t, nullptr, ee,
                                                  E_PAD, HC_DIM, C_DIM, 0);
    }

    // 4) attention logits + segment softmax
    init_softmax_kernel<<<dim3((N_NODES * HEADS + TB - 1) / TB), blk, 0, stream>>>(mx, den);
    edge_logits_kernel<<<dim3(E_EDGES), dim3(HEADS * 32), 0, stream>>>(
        xl, xr, ee, edge_idx, att, logits, mx);
    edge_exp_kernel<<<dim3((E_EDGES * HEADS + TB - 1) / TB), blk, 0, stream>>>(
        logits, mx, edge_idx, ex, den);

    // 5) scatter aggregation into out (= xr storage, now dead) preloaded w/ bias
    {
        long long n1 = (long long)N_PAD * HC_DIM;
        init_out_kernel<<<dim3((unsigned)((n1 + TB - 1) / TB)), blk, 0, stream>>>(out, conv_bias);
        edge_aggregate_kernel<<<dim3(E_EDGES), blk, 0, stream>>>(xl, edge_idx, ex, den, out);
    }

    // 6) decoder: relu(out @ W_d1 + b_d1) via WMMA, then 512->6 head + sigmoid
    {
        long long n4 = (long long)N_PAD * HC_DIM / 4;
        cvt_f16_kernel<<<dim3((unsigned)((n4 + TB - 1) / TB)), blk, 0, stream>>>(out, out_f16, n4);
        dim3 g(N_PAD / 16, C_DIM / 256), b128(128);
        wmma_gemm_kernel<<<g, b128, 0, stream>>>(out_f16, Wd1_t, b_d1, t1,
                                                 N_PAD, C_DIM, HC_DIM, 1);
        decoder2_kernel<<<dim3((N_NODES * 6 + TB - 1) / TB), blk, 0, stream>>>(
            t1, W_d2, b_d2, pred);
    }
    (void)in_sizes; (void)n_in; (void)out_size; (void)ws_size;
}
